// FlowtronTTS_32847909879900
// MI455X (gfx1250) — matrix operand order repaired and checked by reading, not compiled, not run on hardware
//
#include <hip/hip_runtime.h>
#include <hip/hip_bf16.h>

typedef __attribute__((ext_vector_type(16))) __bf16 v16bf;
typedef __attribute__((ext_vector_type(8)))  __bf16 v8bf;
typedef __attribute__((ext_vector_type(8)))  float  v8f;

// ---------------- problem dims ----------------
#define HDIM 1024
#define ADIM 128
#define MDIM 80
#define MPAD 96          // mel input padded to multiple of 32 (zeros)
#define TTD  192
#define TD   512
#define EDIM 640
#define CATD 1152        // H + A

// persistent grid
#define NBLK 64
#define TPB  128         // 4 wave32 per block -> 256 waves total

// ---------------- workspace layout ----------------
#define OFF_STATE  256                       // bytes; [0..256) = barrier ctrl
// float indices inside state block
#define S_X    0                              // 96  (prev-y, zero padded)
#define S_HA   96
#define S_CA   (96 + 1024)
#define S_H1   (96 + 2048)
#define S_C1   (96 + 3072)
#define S_H2   (96 + 4096)
#define S_C2   (96 + 5120)
#define S_Z    (96 + 6144)                    // 4096 gate pre-activations
#define S_HDV  (96 + 10240)                   // 1024
#define S_CTX  (96 + 11264)                   // 128
#define S_BP   (96 + 11392)                   // 160 (b_p | log_s)
#define OFF_SEQ   (OFF_STATE + 64*1024)       // [512][80] f32 working sequence
#define OFF_KV    (OFF_SEQ + 256*1024)        // [2 flow][2 kv][192][128] f32
#define OFF_WTS   (OFF_KV + 512*1024)         // bf16 weights start here
// per-flow bf16 element offsets (row-major, K-contiguous)
#define WE_IH_ATT 0
#define WE_HH_ATT (WE_IH_ATT + 4096*MPAD)
#define WE_Q      (WE_HH_ATT + 4096*1024)     // [128][1024] (transposed)
#define WE_IH1    (WE_Q + 128*1024)
#define WE_HH1    (WE_IH1 + 4096*1152)
#define WE_IH2    (WE_HH1 + 4096*1024)
#define WE_HH2    (WE_IH2 + 4096*1024)
#define WE_D      (WE_HH2 + 4096*1024)        // [1024][1024] (transposed)
#define WE_O      (WE_D + 1024*1024)          // [160][1024]  (transposed)
#define WE_FLOW   (WE_O + 160*1024)

// ---------------- helpers ----------------
__device__ __forceinline__ float sigmoidf_(float x) { return 1.f / (1.f + expf(-x)); }

// WGP-scope prefetch (scope field 0): pulls into ALL cache levels including the
// WGP cache. The __builtin_prefetch locality knob only reaches SE/SYS scopes,
// which skip the WGP cache — useless here since weights are already L2 resident.
__device__ __forceinline__ void prefetch_wgp(const void* p) {
  asm volatile("global_prefetch_b8 %0, off" :: "v"(p));
}

__device__ __forceinline__ void gsync(unsigned* ctrl) {
  __threadfence();            // make our global stores device-visible
  __syncthreads();
  if (threadIdx.x == 0) {
    unsigned gen  = __hip_atomic_load(&ctrl[1], __ATOMIC_RELAXED, __HIP_MEMORY_SCOPE_AGENT);
    unsigned prev = __hip_atomic_fetch_add(&ctrl[0], 1u, __ATOMIC_ACQ_REL, __HIP_MEMORY_SCOPE_AGENT);
    if (prev == NBLK - 1u) {
      __hip_atomic_store(&ctrl[0], 0u, __ATOMIC_RELAXED, __HIP_MEMORY_SCOPE_AGENT);
      __hip_atomic_store(&ctrl[1], gen + 1u, __ATOMIC_RELEASE, __HIP_MEMORY_SCOPE_AGENT);
    } else {
      while (__hip_atomic_load(&ctrl[1], __ATOMIC_ACQUIRE, __HIP_MEMORY_SCOPE_AGENT) == gen)
        __builtin_amdgcn_s_sleep(1);
    }
  }
  __syncthreads();
}

// one 16x32 WMMA K-chunk: acc += W[r0:r0+16, k0:k0+32] * x (x replicated to 16 cols)
__device__ __forceinline__ v8f mv_chunk(v8f acc, const __bf16* __restrict__ wrow,
                                        const __bf16* __restrict__ xv,
                                        int k0, int ka, int kb) {
  v8bf a0 = *(const v8bf*)(wrow + k0 + ka);
  v8bf a1 = *(const v8bf*)(wrow + k0 + ka + 16);
  v8bf b0 = *(const v8bf*)(xv + k0 + kb);
  v8bf b1 = *(const v8bf*)(xv + k0 + kb + 16);
  v16bf a = __builtin_shufflevector(a0, a1, 0,1,2,3,4,5,6,7,8,9,10,11,12,13,14,15);
  v16bf b = __builtin_shufflevector(b0, b1, 0,1,2,3,4,5,6,7,8,9,10,11,12,13,14,15);
  prefetch_wgp(wrow + k0 + 512);               // warm L0 ~1KB ahead in this row
  return __builtin_amdgcn_wmma_f32_16x16x32_bf16(false, a, false, b, (short)0, acc,
                                                 false, false);
}

// out[r0 : r0+16] += W[r0:r0+16, 0:C] * x  with two independent wmma chains.
// A tile: lane(l) row = l&15; halves 0..7 -> K = kA..kA+7, halves 8..15 -> K = 16+kA..  (kA = 8*(l>=16))
// B tile: x replicated into all 16 columns; lane l halves 0..7 -> K = kB..kB+7, halves 8..15 -> K = 16+kB..  (kB = 8*(l>=16))
__device__ __forceinline__ v8f mv16(v8f acc, const __bf16* __restrict__ W, int C,
                                    int r0, const __bf16* __restrict__ xv) {
  const int lane = threadIdx.x & 31;
  const int hi   = lane >> 4;
  const int ka   = hi << 3;
  const int kb   = hi << 3;
  const __bf16* wrow = W + (size_t)(r0 + (lane & 15)) * (size_t)C;
  v8f acc1 = {};
  int k0 = 0;
  for (; k0 + 64 <= C; k0 += 64) {
    acc  = mv_chunk(acc,  wrow, xv, k0,      ka, kb);
    acc1 = mv_chunk(acc1, wrow, xv, k0 + 32, ka, kb);
  }
  if (k0 < C) acc = mv_chunk(acc, wrow, xv, k0, ka, kb);
  return acc + acc1;
}

// ---------------- preprocessing kernels ----------------
__global__ void k_cvt_pad(const float* __restrict__ src, __bf16* __restrict__ dst,
                          int R, int C, int Cp) {
  int n = R * Cp;
  for (int i = blockIdx.x * blockDim.x + threadIdx.x; i < n; i += gridDim.x * blockDim.x) {
    int r = i / Cp, c = i - r * Cp;
    dst[i] = (c < C) ? (__bf16)src[(size_t)r * C + c] : (__bf16)0.0f;
  }
}

// src [Rin][Cin] (f32) -> dst [Cin][Rin] (bf16)
__global__ void k_cvt_T(const float* __restrict__ src, __bf16* __restrict__ dst,
                        int Rin, int Cin) {
  int n = Rin * Cin;
  for (int i = blockIdx.x * blockDim.x + threadIdx.x; i < n; i += gridDim.x * blockDim.x) {
    int o = i / Rin, r = i - o * Rin;
    dst[i] = (__bf16)src[(size_t)r * Cin + o];
  }
}

// keys/vals = enc @ {Wk,Wv} for both flows: out [2][2][TT][A]
__global__ void k_encode(const float* __restrict__ emb, const float* __restrict__ spk,
                         const float* __restrict__ Wk, const float* __restrict__ Wv,
                         const int* __restrict__ text, float* __restrict__ kvout) {
  int i = blockIdx.x * blockDim.x + threadIdx.x;
  if (i >= 2 * 2 * TTD * ADIM) return;
  int a  = i & (ADIM - 1);
  int tt = (i >> 7) % TTD;
  int kv = (i / (ADIM * TTD)) & 1;
  int f  = i / (2 * ADIM * TTD);
  const float* W = (kv == 0 ? Wk : Wv) + (size_t)f * EDIM * ADIM;
  const float* erow = emb + (size_t)text[tt] * 512;
  float acc = 0.f;
  for (int e = 0; e < 512; ++e)  acc += erow[e] * W[(size_t)e * ADIM + a];
  for (int e = 0; e < ADIM; ++e) acc += spk[e] * W[(size_t)(512 + e) * ADIM + a];
  kvout[i] = acc;
}

__global__ void k_seq_in(const float* __restrict__ residual, float* __restrict__ seq) {
  int i = blockIdx.x * blockDim.x + threadIdx.x;   // t*MDIM + m
  if (i >= TD * MDIM) return;
  int t = i / MDIM, m = i - t * MDIM;
  seq[i] = residual[(size_t)m * TD + t];
}

__global__ void k_seq_out(const float* __restrict__ seq, float* __restrict__ out) {
  int i = blockIdx.x * blockDim.x + threadIdx.x;   // m*TD + t
  if (i >= TD * MDIM) return;
  int m = i / TD, t = i - m * TD;
  out[i] = seq[(size_t)t * MDIM + m];
}

// ---------------- persistent flow kernel ----------------
__global__ void __launch_bounds__(TPB, 1)
k_flow(char* __restrict__ ws,
       const float* __restrict__ b_att, const float* __restrict__ b1,
       const float* __restrict__ b2, const float* __restrict__ bd,
       const float* __restrict__ bo) {
  unsigned* ctrl = (unsigned*)ws;
  float* st  = (float*)(ws + OFF_STATE);
  float* seq = (float*)(ws + OFF_SEQ);
  float* kvb = (float*)(ws + OFF_KV);
  __bf16* wts = (__bf16*)(ws + OFF_WTS);

  __shared__ __bf16 s_v[2304];   // bf16 staging of input vectors
  __shared__ float  s_f[352];    // q (128) | scores (192) | 1/sum

  const int tid = threadIdx.x;
  const int gt  = blockIdx.x * TPB + tid;
  const int wav = (blockIdx.x << 2) | (tid >> 5);
  const int lane = tid & 31;
  const int wlo  = ((lane & 15) == 0);        // writer lanes 0 and 16
  const int wbase8 = (lane >> 4) << 3;        // +0 or +8 rows

  for (int pass = 0; pass < 2; ++pass) {
    const int fi = pass ? 0 : 1;              // backward flow first (weights[1])
    const __bf16* wf = wts + (size_t)fi * WE_FLOW;
    const float* keys = kvb + (size_t)(fi * 2 + 0) * TTD * ADIM;
    const float* vals = kvb + (size_t)(fi * 2 + 1) * TTD * ADIM;

    for (int i = gt; i < 96 + 6 * 1024; i += NBLK * TPB) st[i] = 0.f;  // zero states
    gsync(ctrl);

    for (int s = 0; s < TD; ++s) {
      const int t = pass ? s : (TD - 1 - s);

      // ---- phase A: z = Wih_att * x_prev + Whh_att * ha + b_att ----
      for (int i = tid; i < MPAD; i += TPB) s_v[i] = (__bf16)st[S_X + i];
      for (int i = tid; i < HDIM; i += TPB) s_v[MPAD + i] = (__bf16)st[S_HA + i];
      __syncthreads();
      {
        int r0 = wav * 16;
        v8f acc = {};
        acc = mv16(acc, wf + WE_IH_ATT, MPAD, r0, s_v);
        acc = mv16(acc, wf + WE_HH_ATT, HDIM, r0, s_v + MPAD);
        if (wlo) {
          int base = r0 + wbase8;
          for (int i = 0; i < 8; ++i)
            st[S_Z + base + i] = acc[i] + b_att[fi * 4096 + base + i];
        }
      }
      gsync(ctrl);

      // ---- att cell pointwise ----
      if (gt < HDIM) {
        float zi = st[S_Z + gt],        zf = st[S_Z + 1024 + gt];
        float zg = st[S_Z + 2048 + gt], zo = st[S_Z + 3072 + gt];
        float c = sigmoidf_(zf) * st[S_CA + gt] + sigmoidf_(zi) * tanhf(zg);
        st[S_CA + gt] = c;
        st[S_HA + gt] = sigmoidf_(zo) * tanhf(c);
      }
      gsync(ctrl);

      // ---- attention (block 0; 128 threads) ----
      if (blockIdx.x == 0) {
        const __bf16* wq = wf + WE_Q;
        int a = tid;
        const __bf16* row = wq + (size_t)a * HDIM;
        float q = 0.f;
        for (int h = 0; h < HDIM; ++h) q += (float)row[h] * st[S_HA + h];
        s_f[a] = q;
        __syncthreads();
        for (int k = tid; k < TTD; k += TPB) {
          const float* kr = keys + (size_t)k * ADIM;
          float sc = 0.f;
          for (int a2 = 0; a2 < ADIM; ++a2) sc += s_f[a2] * kr[a2];
          s_f[128 + k] = sc * 0.08838834764831845f;  // 1/sqrt(128)
        }
        __syncthreads();
        if (tid == 0) {
          float mx = -1e30f;
          for (int k = 0; k < TTD; ++k) mx = fmaxf(mx, s_f[128 + k]);
          float sum = 0.f;
          for (int k = 0; k < TTD; ++k) { float e = expf(s_f[128 + k] - mx); s_f[128 + k] = e; sum += e; }
          s_f[320] = 1.f / sum;
        }
        __syncthreads();
        float cx = 0.f;
        for (int k = 0; k < TTD; ++k) cx += s_f[128 + k] * vals[(size_t)k * ADIM + a];
        st[S_CTX + a] = cx * s_f[320];
      }
      gsync(ctrl);

      // ---- phase C: z = Wih1 * [ha;ctx] + Whh1 * h1 + b1 ----
      for (int i = tid; i < HDIM; i += TPB) s_v[i] = (__bf16)st[S_HA + i];
      for (int i = tid; i < ADIM; i += TPB) s_v[HDIM + i] = (__bf16)st[S_CTX + i];
      for (int i = tid; i < HDIM; i += TPB) s_v[CATD + i] = (__bf16)st[S_H1 + i];
      __syncthreads();
      {
        int r0 = wav * 16;
        v8f acc = {};
        acc = mv16(acc, wf + WE_IH1, CATD, r0, s_v);
        acc = mv16(acc, wf + WE_HH1, HDIM, r0, s_v + CATD);
        if (wlo) {
          int base = r0 + wbase8;
          for (int i = 0; i < 8; ++i)
            st[S_Z + base + i] = acc[i] + b1[fi * 4096 + base + i];
        }
      }
      gsync(ctrl);
      if (gt < HDIM) {
        float zi = st[S_Z + gt],        zf = st[S_Z + 1024 + gt];
        float zg = st[S_Z + 2048 + gt], zo = st[S_Z + 3072 + gt];
        float c = sigmoidf_(zf) * st[S_C1 + gt] + sigmoidf_(zi) * tanhf(zg);
        st[S_C1 + gt] = c;
        st[S_H1 + gt] = sigmoidf_(zo) * tanhf(c);
      }
      gsync(ctrl);

      // ---- phase D: z = Wih2 * h1 + Whh2 * h2 + b2 ----
      for (int i = tid; i < HDIM; i += TPB) s_v[i] = (__bf16)st[S_H1 + i];
      for (int i = tid; i < HDIM; i += TPB) s_v[HDIM + i] = (__bf16)st[S_H2 + i];
      __syncthreads();
      {
        int r0 = wav * 16;
        v8f acc = {};
        acc = mv16(acc, wf + WE_IH2, HDIM, r0, s_v);
        acc = mv16(acc, wf + WE_HH2, HDIM, r0, s_v + HDIM);
        if (wlo) {
          int base = r0 + wbase8;
          for (int i = 0; i < 8; ++i)
            st[S_Z + base + i] = acc[i] + b2[fi * 4096 + base + i];
        }
      }
      gsync(ctrl);
      if (gt < HDIM) {
        float zi = st[S_Z + gt],        zf = st[S_Z + 1024 + gt];
        float zg = st[S_Z + 2048 + gt], zo = st[S_Z + 3072 + gt];
        float c = sigmoidf_(zf) * st[S_C2 + gt] + sigmoidf_(zi) * tanhf(zg);
        st[S_C2 + gt] = c;
        st[S_H2 + gt] = sigmoidf_(zo) * tanhf(c);
      }
      gsync(ctrl);

      // ---- phase E1: hd = tanh(Wd^T * h2 + bd) ----
      for (int i = tid; i < HDIM; i += TPB) s_v[i] = (__bf16)st[S_H2 + i];
      __syncthreads();
      if (wav < 64) {
        int r0 = wav * 16;
        v8f acc = {};
        acc = mv16(acc, wf + WE_D, HDIM, r0, s_v);
        if (wlo) {
          int base = r0 + wbase8;
          for (int i = 0; i < 8; ++i)
            st[S_HDV + base + i] = tanhf(acc[i] + bd[fi * 1024 + base + i]);
        }
      }
      gsync(ctrl);

      // ---- phase E2: [b_p|log_s] = Wo^T * hd + bo ----
      for (int i = tid; i < HDIM; i += TPB) s_v[i] = (__bf16)st[S_HDV + i];
      __syncthreads();
      if (wav < 10) {
        int r0 = wav * 16;
        v8f acc = {};
        acc = mv16(acc, wf + WE_O, HDIM, r0, s_v);
        if (wlo) {
          int base = r0 + wbase8;
          for (int i = 0; i < 8; ++i)
            st[S_BP + base + i] = acc[i] + bo[fi * 160 + base + i];
        }
      }
      gsync(ctrl);

      // ---- inverse affine coupling: y = (x_t - b_p) * exp(-log_s) ----
      if (gt < MDIM) {
        float xv = seq[t * MDIM + gt];
        float y  = (xv - st[S_BP + gt]) * expf(-st[S_BP + 80 + gt]);
        seq[t * MDIM + gt] = y;
        st[S_X + gt] = y;           // carry for next step
      }
      gsync(ctrl);
    }
  }
}

// ---------------- host launch ----------------
extern "C" void kernel_launch(void* const* d_in, const int* in_sizes, int n_in,
                              void* d_out, int out_size, void* d_ws, size_t ws_size,
                              hipStream_t stream) {
  (void)in_sizes; (void)n_in; (void)out_size; (void)ws_size;
  const float* residual = (const float*)d_in[0];
  const float* spk      = (const float*)d_in[1];
  const float* emb      = (const float*)d_in[2];
  const float* Wih_att  = (const float*)d_in[3];
  const float* Whh_att  = (const float*)d_in[4];
  const float* b_att    = (const float*)d_in[5];
  const float* Wq       = (const float*)d_in[6];
  const float* Wk       = (const float*)d_in[7];
  const float* Wv       = (const float*)d_in[8];
  const float* Wih1     = (const float*)d_in[9];
  const float* Whh1     = (const float*)d_in[10];
  const float* b1       = (const float*)d_in[11];
  const float* Wih2     = (const float*)d_in[12];
  const float* Whh2     = (const float*)d_in[13];
  const float* b2       = (const float*)d_in[14];
  const float* Wd       = (const float*)d_in[15];
  const float* bd       = (const float*)d_in[16];
  const float* Wo       = (const float*)d_in[17];
  const float* bo       = (const float*)d_in[18];
  const int*   text     = (const int*)d_in[21];
  float* out = (float*)d_out;

  char* ws = (char*)d_ws;
  __bf16* wts = (__bf16*)(ws + OFF_WTS);
  float* seq  = (float*)(ws + OFF_SEQ);
  float* kvb  = (float*)(ws + OFF_KV);

  hipMemsetAsync(ws, 0, 256, stream);   // barrier control block

  auto gsize = [](long n) { int b = (int)((n + 255) / 256); return b > 32768 ? 32768 : b; };

  // weight conversion (both flows)
  for (int f = 0; f < 2; ++f) {
    __bf16* wfl = wts + (size_t)f * WE_FLOW;
    k_cvt_pad<<<gsize(4096L*MPAD), 256, 0, stream>>>(Wih_att + (size_t)f*4096*80,  wfl + WE_IH_ATT, 4096, 80,  MPAD);
    k_cvt_pad<<<gsize(4096L*1024), 256, 0, stream>>>(Whh_att + (size_t)f*4096*1024, wfl + WE_HH_ATT, 4096, 1024, 1024);
    k_cvt_T  <<<gsize(1024L*128), 256, 0, stream>>>(Wq      + (size_t)f*1024*128,  wfl + WE_Q,      1024, 128);
    k_cvt_pad<<<gsize(4096L*1152), 256, 0, stream>>>(Wih1   + (size_t)f*4096*1152, wfl + WE_IH1,    4096, 1152, 1152);
    k_cvt_pad<<<gsize(4096L*1024), 256, 0, stream>>>(Whh1   + (size_t)f*4096*1024, wfl + WE_HH1,    4096, 1024, 1024);
    k_cvt_pad<<<gsize(4096L*1024), 256, 0, stream>>>(Wih2   + (size_t)f*4096*1024, wfl + WE_IH2,    4096, 1024, 1024);
    k_cvt_pad<<<gsize(4096L*1024), 256, 0, stream>>>(Whh2   + (size_t)f*4096*1024, wfl + WE_HH2,    4096, 1024, 1024);
    k_cvt_T  <<<gsize(1024L*1024), 256, 0, stream>>>(Wd     + (size_t)f*1024*1024, wfl + WE_D,      1024, 1024);
    k_cvt_T  <<<gsize(1024L*160), 256, 0, stream>>>(Wo      + (size_t)f*1024*160,  wfl + WE_O,      1024, 160);
  }

  // encoder keys/vals + working sequence
  k_encode<<<(2*2*TTD*ADIM + 127) / 128, 128, 0, stream>>>(emb, spk, Wk, Wv, text, kvb);
  k_seq_in<<<(TD*MDIM + 255) / 256, 256, 0, stream>>>(residual, seq);

  // persistent cooperative scan: backward flow (weights[1]) then forward (weights[0])
  k_flow<<<NBLK, TPB, 0, stream>>>(ws, b_att, b1, b2, bd, bo);

  // [T][M] -> [1][M][T]
  k_seq_out<<<(TD*MDIM + 255) / 256, 256, 0, stream>>>(seq, out);
}
